// MixtureOfExpertsLayer_23115513987492
// MI455X (gfx1250) — matrix operand order repaired and checked by reading, not compiled, not run on hardware
//
#include <hip/hip_runtime.h>

#define N_TOK 4096
#define DIM   1024
#define NEXP  8
#define HID   4096
#define ODIM  1024

#define BM 128
#define BN 128
#define BK 32
#define LDT (BK + 8)   // padded ushort stride to dodge LDS bank conflicts

typedef __attribute__((ext_vector_type(16))) __bf16 v16bf;
typedef __attribute__((ext_vector_type(2)))  __bf16 v2bf;
typedef __attribute__((ext_vector_type(8)))  float  v8f;

union Frag16 { v16bf v; unsigned int u[8]; };

#if defined(__has_builtin)
#if __has_builtin(__builtin_amdgcn_cvt_pk_bf16_f32)
#define HAVE_CVT_PK_BF16 1
#endif
#endif

// ---- workspace layout (bytes) ----
#define WS_CNT   0          // int[8]
#define WS_OFF   64         // int[9]
#define WS_RUN   128        // int[8]
#define WS_IMP   192        // float[8]
#define WS_TOPE  256                        // int[N*2]
#define WS_TOPG  (WS_TOPE + N_TOK*2*4)      // float[N*2]
#define WS_TIDX  (WS_TOPG + N_TOK*2*4)      // int[N*2] compacted
#define WS_TG    (WS_TIDX + N_TOK*2*4)      // float[N*2] compacted
#define WS_XBF   (WS_TG   + N_TOK*2*4)      // bf16[N*DIM]
#define WS_HBF   (WS_XBF  + (size_t)N_TOK*DIM*2)  // bf16[(N*2)*HID]

// f32 -> bf16, truncating (RTZ). One shift; folds into *_store_d16_hi_b16 on stores.
__device__ __forceinline__ unsigned short f2bf(float f) {
#ifdef HAVE_CVT_PK_BF16
  v2bf p = __builtin_amdgcn_cvt_pk_bf16_f32(f, 0.f);
  return (unsigned short)(__builtin_bit_cast(unsigned int, p) & 0xFFFFu);
#else
  return (unsigned short)(__float_as_uint(f) >> 16);
#endif
}

// two f32 -> packed bf16x2: one v_perm_b32 (bytes [7:6] of hi, [3:2] of lo)
__device__ __forceinline__ unsigned int pack_bf16x2(float lo, float hi) {
#ifdef HAVE_CVT_PK_BF16
  v2bf p = __builtin_amdgcn_cvt_pk_bf16_f32(lo, hi);
  return __builtin_bit_cast(unsigned int, p);
#else
  return __builtin_amdgcn_perm(__float_as_uint(hi), __float_as_uint(lo), 0x07060302u);
#endif
}

// ---------------- zero init ----------------
__global__ void zero_kernel(float* __restrict__ out, int* __restrict__ hdr) {
  size_t i = (size_t)blockIdx.x * blockDim.x + threadIdx.x;
  size_t tot = (size_t)N_TOK * ODIM + 1;
  if (i < tot) out[i] = 0.f;
  if (i < 64) hdr[i] = 0;   // counts, offsets, running, importance
}

// ---------------- gating: logits, top-2 softmax, x->bf16 ----------------
__global__ __launch_bounds__(256) void gating_kernel(
    const float* __restrict__ x, const float* __restrict__ Wg,
    unsigned short* __restrict__ xbf, int* __restrict__ cnt,
    float* __restrict__ imp, int* __restrict__ tope, float* __restrict__ topg)
{
  int n = blockIdx.x;
  int t = threadIdx.x;
  float acc[NEXP];
#pragma unroll
  for (int e = 0; e < NEXP; ++e) acc[e] = 0.f;

#pragma unroll
  for (int i = 0; i < DIM / 256; ++i) {
    int d = t + i * 256;
    float xv = x[(size_t)n * DIM + d];
    xbf[(size_t)n * DIM + d] = f2bf(xv);
#pragma unroll
    for (int e = 0; e < NEXP; ++e) acc[e] += xv * Wg[d * NEXP + e];
  }
  // wave32 reduce
#pragma unroll
  for (int e = 0; e < NEXP; ++e)
    for (int o = 16; o > 0; o >>= 1) acc[e] += __shfl_down(acc[e], o, 32);

  __shared__ float wsum[8][NEXP];
  int wid = t >> 5, lane = t & 31;
  if (lane == 0)
    for (int e = 0; e < NEXP; ++e) wsum[wid][e] = acc[e];
  __syncthreads();

  if (t == 0) {
    float lg[NEXP];
#pragma unroll
    for (int e = 0; e < NEXP; ++e) {
      float s = 0.f;
      for (int w = 0; w < 8; ++w) s += wsum[w][e];
      lg[e] = s;
    }
    int i0 = 0;
    for (int e = 1; e < NEXP; ++e) if (lg[e] > lg[i0]) i0 = e;
    int i1 = (i0 == 0) ? 1 : 0;
    for (int e = 0; e < NEXP; ++e) if (e != i0 && lg[e] > lg[i1]) i1 = e;
    float ex = __expf(lg[i1] - lg[i0]);
    float g0 = 1.f / (1.f + ex);
    float g1 = ex / (1.f + ex);
    tope[n * 2] = i0; tope[n * 2 + 1] = i1;
    topg[n * 2] = g0; topg[n * 2 + 1] = g1;
    atomicAdd(&cnt[i0], 1); atomicAdd(&cnt[i1], 1);
    atomicAdd(&imp[i0], g0); atomicAdd(&imp[i1], g1);
  }
}

// ---------------- offsets + aux loss ----------------
__global__ void offsets_aux_kernel(const int* __restrict__ cnt, int* __restrict__ offs,
                                   const float* __restrict__ imp, float* __restrict__ out)
{
  int o = 0;
  for (int e = 0; e < NEXP; ++e) { offs[e] = o; o += cnt[e]; }
  offs[NEXP] = o;
  float m = 0.f;
  for (int e = 0; e < NEXP; ++e) m += imp[e];
  m *= (1.f / NEXP);
  float v = 0.f;
  for (int e = 0; e < NEXP; ++e) { float d = imp[e] - m; v += d * d; }
  v *= (1.f / NEXP);
  out[(size_t)N_TOK * ODIM] = 0.01f * v / (m * m + 1e-10f);
}

// ---------------- scatter into per-expert compact lists ----------------
__global__ void scatter_kernel(const int* __restrict__ tope, const float* __restrict__ topg,
                               const int* __restrict__ offs, int* __restrict__ running,
                               int* __restrict__ tidx, float* __restrict__ tg)
{
  int a = blockIdx.x * blockDim.x + threadIdx.x;
  if (a >= N_TOK * 2) return;
  int e = tope[a];
  int slot = atomicAdd(&running[e], 1);
  int row = offs[e] + slot;
  tidx[row] = a >> 1;
  tg[row] = topg[a];
}

// ---- WMMA fragment loads from LDS (layouts per ISA 7.12.2, bf16) ----
__device__ __forceinline__ void load_a_frag(const unsigned short* As, int row0, int lane,
                                            Frag16& f) {
  const unsigned short* p = As + (row0 + (lane & 15)) * LDT;
  int hi = (lane >> 4) << 3;            // 0 or 8
#pragma unroll
  for (int v = 0; v < 8; ++v) {
    int k = (v < 4) ? (hi + v * 2) : (16 + hi + (v - 4) * 2);
    f.u[v] = *(const unsigned int*)(p + k);
  }
}
__device__ __forceinline__ void load_b_frag(const unsigned short* Bs, int col0, int lane,
                                            Frag16& f) {
  const unsigned short* p = Bs + (col0 + (lane & 15)) * LDT + ((lane >> 4) << 3);
#pragma unroll
  for (int v = 0; v < 8; ++v) f.u[v] = *(const unsigned int*)(p + ((v < 4) ? (v * 2) : (16 + (v - 4) * 2)));
}

// store a staged 4(k) x 4(n) fp32 quad as transposed bf16 into LDS (4x ds_store_b64)
__device__ __forceinline__ void store_b_quad(unsigned short* Bs, int bnq, int bkq,
                                             const float4& r0, const float4& r1,
                                             const float4& r2, const float4& r3) {
  uint2 s;
  s.x = pack_bf16x2(r0.x, r1.x); s.y = pack_bf16x2(r2.x, r3.x);
  *(uint2*)(Bs + (bnq + 0) * LDT + bkq) = s;
  s.x = pack_bf16x2(r0.y, r1.y); s.y = pack_bf16x2(r2.y, r3.y);
  *(uint2*)(Bs + (bnq + 1) * LDT + bkq) = s;
  s.x = pack_bf16x2(r0.z, r1.z); s.y = pack_bf16x2(r2.z, r3.z);
  *(uint2*)(Bs + (bnq + 2) * LDT + bkq) = s;
  s.x = pack_bf16x2(r0.w, r1.w); s.y = pack_bf16x2(r2.w, r3.w);
  *(uint2*)(Bs + (bnq + 3) * LDT + bkq) = s;
}

// ---------------- expert GEMM 1: h = relu(x_gathered @ W1[e] + b1[e]) ----------------
__global__ __launch_bounds__(256) void ffn1_kernel(
    const unsigned short* __restrict__ xbf, const float* __restrict__ W1,
    const float* __restrict__ b1, const int* __restrict__ cnt, const int* __restrict__ offs,
    const int* __restrict__ tidx, unsigned short* __restrict__ hbf)
{
  int e = blockIdx.z;
  int ne = cnt[e];
  int m0 = blockIdx.x * BM;
  if (m0 >= ne) return;
  int n0 = blockIdx.y * BN;
  int row0 = offs[e];
  const float* W = W1 + (size_t)e * DIM * HID;

  __shared__ unsigned short As[BM * LDT];
  __shared__ unsigned short Bs[BN * LDT];
  __shared__ int toks[BM];

  int t = threadIdx.x;
  if (t < BM) {
    int m = m0 + t;
    toks[t] = (m < ne) ? tidx[row0 + m] : -1;
  }
  __syncthreads();

  int wid = t >> 5, lane = t & 31;
  int wm = wid & 1, wn = wid >> 1;          // 2 (M) x 4 (N) wave grid

  v8f c[4][2];
#pragma unroll
  for (int i = 0; i < 4; ++i)
#pragma unroll
    for (int j = 0; j < 2; ++j) c[i][j] = (v8f){};

  int am = t >> 1, ak = (t & 1) * 16;       // A loader: 128 rows x 32 cols, 32B each
  int bkq = (t & 7) * 4, bnq = (t >> 3) * 4; // B loader: 4(k)x4(n) quad per thread

  // --- prologue: stage tile k0=0 into registers ---
  int tokA = toks[am];
  const unsigned short* arow = (tokA >= 0) ? (xbf + (size_t)tokA * DIM) : xbf;
  uint4 ra0 = make_uint4(0u,0u,0u,0u), ra1 = make_uint4(0u,0u,0u,0u);
  if (tokA >= 0) { ra0 = *(const uint4*)(arow + ak); ra1 = *(const uint4*)(arow + ak + 8); }
  const float* wp = W + (size_t)bkq * HID + n0 + bnq;
  float4 rb0 = *(const float4*)(wp);
  float4 rb1 = *(const float4*)(wp + HID);
  float4 rb2 = *(const float4*)(wp + 2 * HID);
  float4 rb3 = *(const float4*)(wp + 3 * HID);

  for (int k0 = 0; k0 < DIM; k0 += BK) {
    // commit staged tile to LDS
    *(uint4*)(As + am * LDT + ak)     = ra0;
    *(uint4*)(As + am * LDT + ak + 8) = ra1;
    store_b_quad(Bs, bnq, bkq, rb0, rb1, rb2, rb3);
    __syncthreads();

    // issue NEXT tile's global loads so they overlap the WMMA burst
    int kn = k0 + BK;
    if (kn < DIM) {
      if (tokA >= 0) {
        ra0 = *(const uint4*)(arow + kn + ak);
        ra1 = *(const uint4*)(arow + kn + ak + 8);
      }
      const float* wpn = W + (size_t)(kn + bkq) * HID + n0 + bnq;
      rb0 = *(const float4*)(wpn);
      rb1 = *(const float4*)(wpn + HID);
      rb2 = *(const float4*)(wpn + 2 * HID);
      rb3 = *(const float4*)(wpn + 3 * HID);
      if (kn + BK < DIM) __builtin_prefetch(wpn + (size_t)BK * HID, 0, 0);
    }

    Frag16 a[4], b[2];
#pragma unroll
    for (int i = 0; i < 4; ++i) load_a_frag(As, wm * 64 + i * 16, lane, a[i]);
#pragma unroll
    for (int j = 0; j < 2; ++j) load_b_frag(Bs, wn * 32 + j * 16, lane, b[j]);
#pragma unroll
    for (int i = 0; i < 4; ++i)
#pragma unroll
      for (int j = 0; j < 2; ++j)
        c[i][j] = __builtin_amdgcn_wmma_f32_16x16x32_bf16(false, a[i].v, false, b[j].v,
                                                          (short)0, c[i][j], false, false);
    __syncthreads();
  }

  int nloc = lane & 15;
  int mhi = (lane >> 4) * 8;
#pragma unroll
  for (int j = 0; j < 2; ++j) {
    int gn = n0 + wn * 32 + j * 16 + nloc;
    float bias = b1[e * HID + gn];
#pragma unroll
    for (int i = 0; i < 4; ++i) {
#pragma unroll
      for (int r = 0; r < 8; ++r) {
        int m = m0 + wm * 64 + i * 16 + mhi + r;
        if (m < ne) {
          float v = c[i][j][r] + bias;
          v = v > 0.f ? v : 0.f;
          hbf[(size_t)(row0 + m) * HID + gn] = f2bf(v);
        }
      }
    }
  }
}

// ---------------- expert GEMM 2: out += gate * (h @ W2[e] + b2[e]) ----------------
__global__ __launch_bounds__(256) void ffn2_kernel(
    const unsigned short* __restrict__ hbf, const float* __restrict__ W2,
    const float* __restrict__ b2, const int* __restrict__ cnt, const int* __restrict__ offs,
    const int* __restrict__ tidx, const float* __restrict__ tg, float* __restrict__ out)
{
  int e = blockIdx.z;
  int ne = cnt[e];
  int m0 = blockIdx.x * BM;
  if (m0 >= ne) return;
  int n0 = blockIdx.y * BN;
  int row0 = offs[e];
  const float* W = W2 + (size_t)e * HID * ODIM;

  __shared__ unsigned short As[BM * LDT];
  __shared__ unsigned short Bs[BN * LDT];
  __shared__ int toks[BM];
  __shared__ float gts[BM];

  int t = threadIdx.x;
  if (t < BM) {
    int m = m0 + t;
    toks[t] = (m < ne) ? tidx[row0 + m] : -1;
    gts[t]  = (m < ne) ? tg[row0 + m] : 0.f;
  }
  __syncthreads();

  int wid = t >> 5, lane = t & 31;
  int wm = wid & 1, wn = wid >> 1;

  v8f c[4][2];
#pragma unroll
  for (int i = 0; i < 4; ++i)
#pragma unroll
    for (int j = 0; j < 2; ++j) c[i][j] = (v8f){};

  int am = t >> 1, ak = (t & 1) * 16;
  int bkq = (t & 7) * 4, bnq = (t >> 3) * 4;
  bool avalid = (m0 + am) < ne;
  const unsigned short* arow = hbf + (size_t)(row0 + (avalid ? (m0 + am) : 0)) * HID;

  uint4 ra0 = make_uint4(0u,0u,0u,0u), ra1 = make_uint4(0u,0u,0u,0u);
  if (avalid) { ra0 = *(const uint4*)(arow + ak); ra1 = *(const uint4*)(arow + ak + 8); }
  const float* wp = W + (size_t)bkq * ODIM + n0 + bnq;
  float4 rb0 = *(const float4*)(wp);
  float4 rb1 = *(const float4*)(wp + ODIM);
  float4 rb2 = *(const float4*)(wp + 2 * ODIM);
  float4 rb3 = *(const float4*)(wp + 3 * ODIM);

  for (int k0 = 0; k0 < HID; k0 += BK) {
    *(uint4*)(As + am * LDT + ak)     = ra0;
    *(uint4*)(As + am * LDT + ak + 8) = ra1;
    store_b_quad(Bs, bnq, bkq, rb0, rb1, rb2, rb3);
    __syncthreads();

    int kn = k0 + BK;
    if (kn < HID) {
      if (avalid) {
        ra0 = *(const uint4*)(arow + kn + ak);
        ra1 = *(const uint4*)(arow + kn + ak + 8);
      }
      const float* wpn = W + (size_t)(kn + bkq) * ODIM + n0 + bnq;
      rb0 = *(const float4*)(wpn);
      rb1 = *(const float4*)(wpn + ODIM);
      rb2 = *(const float4*)(wpn + 2 * ODIM);
      rb3 = *(const float4*)(wpn + 3 * ODIM);
      if (kn + BK < HID) __builtin_prefetch(wpn + (size_t)BK * ODIM, 0, 0);
    }

    Frag16 a[4], b[2];
#pragma unroll
    for (int i = 0; i < 4; ++i) load_a_frag(As, wm * 64 + i * 16, lane, a[i]);
#pragma unroll
    for (int j = 0; j < 2; ++j) load_b_frag(Bs, wn * 32 + j * 16, lane, b[j]);
#pragma unroll
    for (int i = 0; i < 4; ++i)
#pragma unroll
      for (int j = 0; j < 2; ++j)
        c[i][j] = __builtin_amdgcn_wmma_f32_16x16x32_bf16(false, a[i].v, false, b[j].v,
                                                          (short)0, c[i][j], false, false);
    __syncthreads();
  }

  int nloc = lane & 15;
  int mhi = (lane >> 4) * 8;
#pragma unroll
  for (int j = 0; j < 2; ++j) {
    int gn = n0 + wn * 32 + j * 16 + nloc;
    float bias = b2[e * ODIM + gn];
#pragma unroll
    for (int i = 0; i < 4; ++i) {
#pragma unroll
      for (int r = 0; r < 8; ++r) {
        int ml = wm * 64 + i * 16 + mhi + r;
        int m = m0 + ml;
        if (m < ne) {
          int tok = toks[ml];
          float g = gts[ml];
          atomicAdd(&out[(size_t)tok * ODIM + gn], g * (c[i][j][r] + bias));
        }
      }
    }
  }
}

extern "C" void kernel_launch(void* const* d_in, const int* in_sizes, int n_in,
                              void* d_out, int out_size, void* d_ws, size_t ws_size,
                              hipStream_t stream) {
  const float* x  = (const float*)d_in[0];
  const float* Wg = (const float*)d_in[1];
  const float* W1 = (const float*)d_in[2];
  const float* b1 = (const float*)d_in[3];
  const float* W2 = (const float*)d_in[4];
  const float* b2 = (const float*)d_in[5];
  float* out = (float*)d_out;
  char* ws = (char*)d_ws;

  int*   cnt  = (int*)(ws + WS_CNT);
  int*   offs = (int*)(ws + WS_OFF);
  int*   run  = (int*)(ws + WS_RUN);
  float* imp  = (float*)(ws + WS_IMP);
  int*   tope = (int*)(ws + WS_TOPE);
  float* topg = (float*)(ws + WS_TOPG);
  int*   tidx = (int*)(ws + WS_TIDX);
  float* tgv  = (float*)(ws + WS_TG);
  unsigned short* xbf = (unsigned short*)(ws + WS_XBF);
  unsigned short* hbf = (unsigned short*)(ws + WS_HBF);

  size_t tot = (size_t)N_TOK * ODIM + 1;
  zero_kernel<<<dim3((unsigned)((tot + 255) / 256)), dim3(256), 0, stream>>>(out, (int*)ws);
  gating_kernel<<<dim3(N_TOK), dim3(256), 0, stream>>>(x, Wg, xbf, cnt, imp, tope, topg);
  offsets_aux_kernel<<<dim3(1), dim3(1), 0, stream>>>(cnt, offs, imp, out);
  scatter_kernel<<<dim3(N_TOK * 2 / 256), dim3(256), 0, stream>>>(tope, topg, offs, run, tidx, tgv);
  ffn1_kernel<<<dim3(N_TOK / BM, HID / BN, NEXP), dim3(256), 0, stream>>>(xbf, W1, b1, cnt, offs, tidx, hbf);
  ffn2_kernel<<<dim3(N_TOK / BM, ODIM / BN, NEXP), dim3(256), 0, stream>>>(hbf, W2, b2, cnt, offs, tidx, tgv, out);
}
